// HungarianMatcher_27839978012859
// MI455X (gfx1250) — matrix operand order repaired and checked by reading, not compiled, not run on hardware
//
#include <hip/hip_runtime.h>

// ---------------------------------------------------------------------------
// Mask3D-style Hungarian matcher cost matrix on MI455X (gfx1250, wave32).
//
// Roofline: outputs_mask is 419 MB streamed once => ~18us floor @ 23.3 TB/s.
// qn,mn->qm einsums vs a one-hot target matrix -> v_wmma_f32_16x16x32_f16.
// Two adjacent 16-query tiles per block share each chunk's one-hot B build
// and label fetch; the kernel is specialized at compile time on 1-vs-2 tiles
// so the inner loop carries no predicate. All four one-hot B fragments are
// materialized before the 16-WMMA block (four simultaneously-live register
// sets -> no B-operand WAR hazard, no v_nop padding). A-tiles (32 points x
// 32 queries, row stride Q*4B) are staged into LDS by the Tensor Data Mover
// with TENSORcnt double buffering.
// ---------------------------------------------------------------------------

typedef __attribute__((ext_vector_type(16))) _Float16     v16h;
typedef __attribute__((ext_vector_type(8)))  float        v8f;
typedef __attribute__((ext_vector_type(4)))  unsigned int u32x4;
typedef __attribute__((ext_vector_type(8)))  int          i32x8;
typedef __attribute__((ext_vector_type(4)))  int          i32x4;

#define B_      4
#define N_      65536
#define Q_      400
#define NC_     20
#define M_      64
#define QT_     25            // Q / 16 query tiles
#define SPLIT_  8             // N-range splits per q-pair (occupancy)
#define NSEG_   (N_ / SPLIT_) // 8192 points per split
#define CHUNKS_ (NSEG_ / 32)  // 256 chunks of 32 points
#define THREADS_ 128
#define WAVES_  4             // waves per block (wave32)

#define TILE_BYTES  4096      // 32 rows x 32 floats
#define SMEM_BYTES  33792     // max(tiles 32KB, red 32KB + sred 1KB)

#if __has_builtin(__builtin_amdgcn_tensor_load_to_lds)
#define USE_TDM 1
#else
#define USE_TDM 0
#endif

// One TDM 2D tile load: 32 lines x TW floats, line stride Q floats.
// D# bitfields per CDNA5 ISA ch.8: group0 = {count=1, lds_addr, global_addr,
// type=2}; group1 = {data_size=4B, tensor_dim0=Q, tensor_dim1=N, tile=TW x 32,
// tensor_dim0_stride=Q}; groups 2/3 zero (<=2D tensor). Wave-uniform args.
__device__ __forceinline__ void tdm_load_tile(unsigned lds_off, const float* gp, int tw) {
#if USE_TDM
  unsigned long long ga = (unsigned long long)(const void*)gp;
  u32x4 g0;
  g0[0] = 1u;                                             // count=1, user mode
  g0[1] = lds_off;                                        // LDS byte address
  g0[2] = (unsigned)ga;                                   // global_addr[31:0]
  g0[3] = ((unsigned)(ga >> 32) & 0x01FFFFFFu)            // global_addr[56:32]
          | 0x80000000u;                                  // type=2 ("image")
  i32x8 g1;
  g1[0] = 2 << 16;                                        // data_size=2 (4 bytes)
  g1[1] = (int)((unsigned)Q_ << 16);                      // tensor_dim0 = Q (lo16)
  g1[2] = 0;                                              // dim0 hi=0, dim1 lo16=0
  g1[3] = (tw << 16) | (int)(N_ >> 16);                   // tile_dim0=tw | dim1 hi16
  g1[4] = 32;                                             // tile_dim1 = 32 lines
  g1[5] = Q_;                                             // tensor_dim0_stride (lo32)
  g1[6] = 0;
  g1[7] = 0;
  i32x4 z4 = {0, 0, 0, 0};
#if __clang_major__ >= 23
  i32x8 z8 = {0, 0, 0, 0, 0, 0, 0, 0};
  __builtin_amdgcn_tensor_load_to_lds(g0, g1, z4, z4, z8, 0);
#else
  __builtin_amdgcn_tensor_load_to_lds(g0, g1, z4, z4, 0);
#endif
#endif
}

// One-hot B fragment for m-tile mt. Contiguous k order used here: lanes 0-15 col=lane hold
// K=0..15, lanes 16-31 hold K=16..31 (instk[] already half-adjusted).
__device__ __forceinline__ v16h build_onehot(const int* instk, int ql, int mt) {
  int mtgt = mt * 16 + ql;
  v16h bh;
  #pragma unroll
  for (int t = 0; t < 16; ++t)
    bh[t] = (instk[t] == mtgt) ? (_Float16)1.0f : (_Float16)0.0f;
  return bh;
}

// ---- kernel 0: zero the tiny per-instance stats (d_ws is poisoned) --------
__global__ void k_init(int* counts, int* segmax) {
  int i = blockIdx.x * blockDim.x + threadIdx.x;
  if (i < B_ * M_) { counts[i] = 0; segmax[i] = 0; }
}

// ---- kernel 1: per-instance point count + max segment label ---------------
// LDS histogram per block (fast ds atomics), then 64 global atomics/block.
__global__ __launch_bounds__(256)
void k_labels(const int* __restrict__ inst, const int* __restrict__ seg,
              int* counts, int* segmax) {
  __shared__ int hcnt[M_];
  __shared__ int hmax[M_];
  int t = threadIdx.x;
  if (t < M_) { hcnt[t] = 0; hmax[t] = 0; }
  __syncthreads();
  int i = blockIdx.x * blockDim.x + t;   // exactly B*N threads
  int b = i / N_;                        // uniform per block (N % 256 == 0)
  atomicAdd(&hcnt[inst[i]], 1);
  atomicMax(&hmax[inst[i]], seg[i]);     // matches max(tgt*seg): seg>=0, empty->0
  __syncthreads();
  if (t < M_) {
    if (hcnt[t] > 0) atomicAdd(&counts[b * M_ + t], hcnt[t]);
    atomicMax(&segmax[b * M_ + t], hmax[t]);
  }
}

// ---- kernel 2: TDM-staged WMMA segmented reduction over points ------------
// HAS2: block owns two adjacent q-tiles (compile-time; no inner predicates).
// QPN / QPOFF: pairs per launch and pair-index offset (grid decode constants).
template <bool HAS2, int QPN, int QPOFF>
__global__ __launch_bounds__(THREADS_)
void k_main(const float* __restrict__ mask, const int* __restrict__ inst,
            float* __restrict__ partial,    // [B][QT][SPLIT][2][16][64]
            float* __restrict__ partialq) { // [B][QT][SPLIT][2][16]
  extern __shared__ char smem[];            // tiles (32KB) overlap red (33KB)

  int blk = blockIdx.x;
  int b   = blk / (QPN * SPLIT_);
  int rm  = blk % (QPN * SPLIT_);
  int qp  = QPOFF + rm / SPLIT_;
  int sp  = rm % SPLIT_;
  int qt0 = qp * 2, qt1 = qt0 + 1;
  constexpr int TW = HAS2 ? 32 : 16;        // TDM tile width in floats

  int w    = threadIdx.x >> 5;
  int lane = threadIdx.x & 31;
  int half = lane >> 4;
  int ql   = lane & 15;

  const float* mb = mask + (size_t)b * N_ * Q_;
  const int*   ib = inst + (size_t)b * N_;
  long nbase = (long)sp * NSEG_;

  v8f accx0[4] = {}, accs0[4] = {};         // tile0: Xsum / SigSum, 4 m-tiles
  v8f accx1[4] = {}, accs1[4] = {};         // tile1 (dead code if !HAS2)
  float sp0 = 0.f, sg0 = 0.f, sp1 = 0.f, sg1 = 0.f;

  const unsigned tile_base = (unsigned)(w * 2 * TILE_BYTES);

  // prologue: prefetch first chunk's tile into buffer 0
  tdm_load_tile(tile_base, mb + (size_t)(nbase + (long)w * 32) * Q_ + qt0 * 16, TW);

  for (int c = w; c < CHUNKS_; c += WAVES_) {
    long n0  = nbase + (long)c * 32;
    int  buf = ((c - w) / WAVES_) & 1;
    int  cn  = c + WAVES_;
    if (cn < CHUNKS_) {       // uniform: CHUNKS_ % WAVES_ == 0
      tdm_load_tile(tile_base + (unsigned)((buf ^ 1) * TILE_BYTES),
                    mb + (size_t)(nbase + (long)cn * 32) * Q_ + qt0 * 16, TW);
#if USE_TDM
      __builtin_amdgcn_s_wait_tensorcnt((short)1);  // current buffer landed
#endif
    } else {
#if USE_TDM
      __builtin_amdgcn_s_wait_tensorcnt((short)0);
#endif
    }
    const float* tile = (const float*)(smem + tile_base + buf * TILE_BYTES);

    // 16 instance labels for this lane's B column: consecutive ints,
    // half-wave-uniform address -> 4x global_load_b128 (cache broadcast).
    // Shared by both q-tiles.
    int4 iv0 = *(const int4*)&ib[n0 + half * 16 + 0];
    int4 iv1 = *(const int4*)&ib[n0 + half * 16 + 4];
    int4 iv2 = *(const int4*)&ib[n0 + half * 16 + 8];
    int4 iv3 = *(const int4*)&ib[n0 + half * 16 + 12];
    int instk[16] = { iv0.x, iv0.y, iv0.z, iv0.w, iv1.x, iv1.y, iv1.z, iv1.w,
                      iv2.x, iv2.y, iv2.z, iv2.w, iv3.x, iv3.y, iv3.z, iv3.w };

    // A fragments (both tiles). Contiguous k order used here: lane half h,
    // element t holds K = h*16+t (matches B), row = lane&15.
    v16h ax0, as0, ax1, as1;
    #pragma unroll
    for (int t = 0; t < 16; ++t) {
      int k = half * 16 + t;
      float x0 = USE_TDM ? tile[k * TW + ql]
                         : mb[(size_t)(n0 + k) * Q_ + qt0 * 16 + ql];
      // raw v_exp_f32 / v_log_f32 / v_rcp_f32 (no denorm fixups; dn >= 1)
      float e0  = __builtin_amdgcn_exp2f(x0 * -1.44269504f);
      float dn0 = 1.0f + e0;
      float s0  = __builtin_amdgcn_rcpf(dn0);                        // sigmoid
      sp0 += fmaf(0.69314718f, __builtin_amdgcn_logf(dn0), x0);      // softplus
      sg0 += s0;
      ax0[t] = (_Float16)x0;
      as0[t] = (_Float16)s0;
      if constexpr (HAS2) {
        float x1 = USE_TDM ? tile[k * TW + 16 + ql]
                           : mb[(size_t)(n0 + k) * Q_ + qt1 * 16 + ql];
        float e1  = __builtin_amdgcn_exp2f(x1 * -1.44269504f);
        float dn1 = 1.0f + e1;
        float s1  = __builtin_amdgcn_rcpf(dn1);
        sp1 += fmaf(0.69314718f, __builtin_amdgcn_logf(dn1), x1);
        sg1 += s1;
        ax1[t] = (_Float16)x1;
        as1[t] = (_Float16)s1;
      }
    }

    // Materialize all four one-hot B fragments first: four simultaneously
    // live register sets cannot be coalesced, so the WMMA block below has no
    // B-operand WAR hazard (cndmask->WMMA is a plain interlocked RAW), and
    // the next chunk's builds are separated from these WMMAs by ~250 A-math
    // instructions -> no v_nop padding anywhere.
    v16h b0 = build_onehot(instk, ql, 0);
    v16h b1 = build_onehot(instk, ql, 1);
    v16h b2 = build_onehot(instk, ql, 2);
    v16h b3 = build_onehot(instk, ql, 3);
    v16h bh[4] = { b0, b1, b2, b3 };

    #pragma unroll
    for (int mt = 0; mt < 4; ++mt) {
      accx0[mt] = __builtin_amdgcn_wmma_f32_16x16x32_f16(
          false, ax0, false, bh[mt], (short)0, accx0[mt], false, false);
      accs0[mt] = __builtin_amdgcn_wmma_f32_16x16x32_f16(
          false, as0, false, bh[mt], (short)0, accs0[mt], false, false);
      if constexpr (HAS2) {
        accx1[mt] = __builtin_amdgcn_wmma_f32_16x16x32_f16(
            false, ax1, false, bh[mt], (short)0, accx1[mt], false, false);
        accs1[mt] = __builtin_amdgcn_wmma_f32_16x16x32_f16(
            false, as1, false, bh[mt], (short)0, accs1[mt], false, false);
      }
    }
  }

  __syncthreads();  // tile epoch ends; LDS region is reused for reduction

  // ---- cross-wave tree reduction in LDS (4 -> 2 -> 1) ---------------------
  float* red  = (float*)smem;              // [2][2][4][2][8][32] = 32 KB
  float* sred = (float*)(smem + 32768);    // [2][2][2][32]       = 1 KB
  #define RIDX(s, tl, mt, qy, r, l) \
    ((((((s) * 2 + (tl)) * 4 + (mt)) * 2 + (qy)) * 8 + (r)) * 32 + (l))
  #define SIDX(s, tl, qy, l) ((((s) * 2 + (tl)) * 2 + (qy)) * 32 + (l))
  for (int off = 2; off >= 1; off >>= 1) {
    if (w >= off && w < 2 * off) {
      #pragma unroll
      for (int mt = 0; mt < 4; ++mt)
        #pragma unroll
        for (int r = 0; r < 8; ++r) {
          red[RIDX(w - off, 0, mt, 0, r, lane)] = accx0[mt][r];
          red[RIDX(w - off, 0, mt, 1, r, lane)] = accs0[mt][r];
          if constexpr (HAS2) {
            red[RIDX(w - off, 1, mt, 0, r, lane)] = accx1[mt][r];
            red[RIDX(w - off, 1, mt, 1, r, lane)] = accs1[mt][r];
          }
        }
      sred[SIDX(w - off, 0, 0, lane)] = sp0;
      sred[SIDX(w - off, 0, 1, lane)] = sg0;
      if constexpr (HAS2) {
        sred[SIDX(w - off, 1, 0, lane)] = sp1;
        sred[SIDX(w - off, 1, 1, lane)] = sg1;
      }
    }
    __syncthreads();
    if (w < off) {
      #pragma unroll
      for (int mt = 0; mt < 4; ++mt)
        #pragma unroll
        for (int r = 0; r < 8; ++r) {
          accx0[mt][r] += red[RIDX(w, 0, mt, 0, r, lane)];
          accs0[mt][r] += red[RIDX(w, 0, mt, 1, r, lane)];
          if constexpr (HAS2) {
            accx1[mt][r] += red[RIDX(w, 1, mt, 0, r, lane)];
            accs1[mt][r] += red[RIDX(w, 1, mt, 1, r, lane)];
          }
        }
      sp0 += sred[SIDX(w, 0, 0, lane)];
      sg0 += sred[SIDX(w, 0, 1, lane)];
      if constexpr (HAS2) {
        sp1 += sred[SIDX(w, 1, 0, lane)];
        sg1 += sred[SIDX(w, 1, 1, lane)];
      }
    }
    __syncthreads();
  }

  // ---- wave 0 writes this split's partial tiles to the workspace ----------
  if (w == 0) {
    // C/D layout: VGPR r = row q=r (lanes 0-15) / q=r+8 (lanes 16-31), col=lane&15
    int mcol = lane & 15;
    #pragma unroll
    for (int tl = 0; tl < (HAS2 ? 2 : 1); ++tl) {
      int qt = tl ? qt1 : qt0;
      float* pp = partial + (((size_t)(b * QT_ + qt) * SPLIT_ + sp) * 2) * (16 * 64);
      #pragma unroll
      for (int mt = 0; mt < 4; ++mt)
        #pragma unroll
        for (int r = 0; r < 8; ++r) {
          int qrow = (lane < 16) ? r : (r + 8);
          float vx = tl ? accx1[mt][r] : accx0[mt][r];
          float vs = tl ? accs1[mt][r] : accs0[mt][r];
          pp[0 * 1024 + qrow * 64 + mt * 16 + mcol] = vx;
          pp[1 * 1024 + qrow * 64 + mt * 16 + mcol] = vs;
        }
      float sp_ = tl ? sp1 : sp0;
      float sg_ = tl ? sg1 : sg0;
      float tsp = sp_ + __shfl_xor(sp_, 16, 32);     // fold the two halves
      float tsg = sg_ + __shfl_xor(sg_, 16, 32);
      if (lane < 16) {
        float* pq = partialq + (((size_t)(b * QT_ + qt) * SPLIT_ + sp) * 2) * 16;
        pq[lane]      = tsp;   // S_neg partial  (sum softplus)
        pq[16 + lane] = tsg;   // SigTot partial (sum sigmoid)
      }
    }
  }
}

// ---- kernel 3: combine partials + class softmax into the cost matrix ------
__global__ void k_combine(const float* __restrict__ cls, const int* __restrict__ counts,
                          const int* __restrict__ segmax,
                          const float* __restrict__ partial,
                          const float* __restrict__ partialq,
                          float* __restrict__ out) {
  int i = blockIdx.x * blockDim.x + threadIdx.x;
  if (i >= B_ * Q_ * M_) return;
  int m = i % M_;
  int q = (i / M_) % Q_;
  int b = i / (Q_ * M_);
  int qt = q >> 4, qr = q & 15;

  float Xs = 0.f, Ss = 0.f, Sneg = 0.f, Stot = 0.f;
  #pragma unroll
  for (int sp = 0; sp < SPLIT_; ++sp) {
    const float* pp = partial  + (((size_t)(b * QT_ + qt) * SPLIT_ + sp) * 2) * 1024;
    const float* pq = partialq + (((size_t)(b * QT_ + qt) * SPLIT_ + sp) * 2) * 16;
    Xs   += pp[qr * 64 + m];
    Ss   += pp[1024 + qr * 64 + m];
    Sneg += pq[qr];
    Stot += pq[16 + qr];
  }

  // softmax(cls[b,q])[segmax[b,m]]  (raw hardware transcendentals)
  const float* crow = cls + ((size_t)b * Q_ + q) * NC_;
  float mx = crow[0];
  #pragma unroll
  for (int c2 = 1; c2 < NC_; ++c2) mx = fmaxf(mx, crow[c2]);
  float se = 0.f;
  #pragma unroll
  for (int c2 = 0; c2 < NC_; ++c2)
    se += __builtin_amdgcn_exp2f((crow[c2] - mx) * 1.44269504f);
  int   cm   = segmax[b * M_ + m];
  float prob = __builtin_amdgcn_exp2f((crow[cm] - mx) * 1.44269504f)
             * __builtin_amdgcn_rcpf(se);

  float cnt = (float)counts[b * M_ + m];
  const float inv_hw = 1.0f / (float)N_;
  float cost_mask  = (Sneg - Xs) * inv_hw;                        // softplus identity
  float cost_dice  = 1.0f - (2.0f * Ss + 1.0f) / (Stot + cnt + 1.0f);
  float cost_class = 1.0f - prob;
  out[i] = cost_mask + cost_class + cost_dice;
}

// ---------------------------------------------------------------------------
extern "C" void kernel_launch(void* const* d_in, const int* in_sizes, int n_in,
                              void* d_out, int out_size, void* d_ws, size_t ws_size,
                              hipStream_t stream) {
  const float* mask = (const float*)d_in[0];   // [B,N,Q]
  const float* cls  = (const float*)d_in[1];   // [B,Q,NC]
  const int*   inst = (const int*)  d_in[2];   // [B,N]
  const int*   seg  = (const int*)  d_in[3];   // [B,N]
  float* out = (float*)d_out;                  // [B,Q,M]

  // workspace carve-up (~6.7 MB total)
  char* ws = (char*)d_ws;
  int* counts = (int*)ws;                 ws += (size_t)B_ * M_ * sizeof(int);
  int* segm   = (int*)ws;                 ws += (size_t)B_ * M_ * sizeof(int);
  float* partialq = (float*)ws;           ws += (size_t)B_ * QT_ * SPLIT_ * 2 * 16 * sizeof(float);
  float* partial  = (float*)ws;

  k_init   <<<(B_ * M_ + 255) / 256, 256, 0, stream>>>(counts, segm);
  k_labels <<<(B_ * N_) / 256,       256, 0, stream>>>(inst, seg, counts, segm);
  // pairs 0..11: two q-tiles per block (unconditional dual-tile inner loop)
  k_main<true, 12, 0><<<B_ * 12 * SPLIT_, THREADS_, SMEM_BYTES, stream>>>(
      mask, inst, partial, partialq);
  // pair 12 (qt = 24): single-tile variant, 16-wide TDM tile, <256 VGPRs
  k_main<false, 1, 12><<<B_ * 1 * SPLIT_, THREADS_, SMEM_BYTES, stream>>>(
      mask, inst, partial, partialq);
  k_combine<<<(B_ * Q_ * M_ + 255) / 256, 256, 0, stream>>>(cls, counts, segm,
                                                            partial, partialq, out);
}
